// MultiHeadAttention_34084860461678
// MI455X (gfx1250) — matrix operand order repaired and checked by reading, not compiled, not run on hardware
//
#include <hip/hip_runtime.h>
#include <hip/hip_bf16.h>

// ---------------------------------------------------------------------------
// MI455X (gfx1250) fused multi-head attention, wave32 + v_wmma_f32_16x16x32_bf16
// ---------------------------------------------------------------------------

typedef __attribute__((ext_vector_type(16))) __bf16        v16bf;
typedef __attribute__((ext_vector_type(8)))  float         v8f;
typedef __attribute__((ext_vector_type(8)))  unsigned int  v8u;

#define EMB   1024
#define NH    16
#define HD    64
#define BATCH 2
#define SEQ   2048

// ---- fp32 -> bf16 via v_perm_b32 (1 perm per pair; +0x8000 = nearest) ------
__device__ __forceinline__ unsigned int pack2bf(float lo, float hi) {
  const unsigned int a = __builtin_bit_cast(unsigned int, lo) + 0x8000u;
  const unsigned int b = __builtin_bit_cast(unsigned int, hi) + 0x8000u;
  // bytes: [0]=a[2] [1]=a[3] [2]=b[2] [3]=b[3]  (selects 0-3 from src1=a)
  return __builtin_amdgcn_perm(b, a, 0x07060302u);
}
__device__ __forceinline__ unsigned short f2bf(float f) {
  return (unsigned short)((__builtin_bit_cast(unsigned int, f) + 0x8000u) >> 16);
}

// A-matrix 16-bit layout (ISA 7.12.2): VGPR j holds K-pair; lanes 16-31 are +8.
__device__ __forceinline__ int kpairA(int j, int h) {
  return ((j < 4) ? (2 * j) : (16 + 2 * (j - 4))) + 8 * h;
}
__device__ __forceinline__ v8f wmma_bf16(v16bf a, v16bf b, v8f c) {
  return __builtin_amdgcn_wmma_f32_16x16x32_bf16(false, a, false, b,
                                                 (short)0, c, false, false);
}
__device__ __forceinline__ float redmax16(float v) {
  v = fmaxf(v, __shfl_xor(v, 1, 16));
  v = fmaxf(v, __shfl_xor(v, 2, 16));
  v = fmaxf(v, __shfl_xor(v, 4, 16));
  v = fmaxf(v, __shfl_xor(v, 8, 16));
  return v;
}
__device__ __forceinline__ float redsum16(float v) {
  v += __shfl_xor(v, 1, 16);
  v += __shfl_xor(v, 2, 16);
  v += __shfl_xor(v, 4, 16);
  v += __shfl_xor(v, 8, 16);
  return v;
}

// ---------------------------------------------------------------------------
// GEMM with bias: C[M,N] = A[M,K] @ B[K,N] + bias[N]   (fp32 in, bf16 WMMA)
// Block tile 128x128, BK=32. 8 waves, wave grid 4(M) x 2(N), wave tile 32x64.
// ---------------------------------------------------------------------------
#define BM 128
#define BN 128
#define BK 32
#define LDA_PAD 40   // 32 + 8 ushorts: frag loads hit 16 distinct banks

__global__ __launch_bounds__(256) void gemm_bias_wmma_k(
    const float* __restrict__ A, const float* __restrict__ Bm,
    const float* __restrict__ bias, float* __restrict__ C,
    int M, int N, int K) {
  __shared__ unsigned short Asl[BM][LDA_PAD];
  __shared__ unsigned short Bsl[BN][LDA_PAD];   // stored N-major (transposed)

  const int tid  = threadIdx.x;
  const int wave = tid >> 5;
  const int lane = tid & 31;
  const int lm = lane & 15, lh = lane >> 4;
  const int m0 = blockIdx.x * BM;
  const int n0 = blockIdx.y * BN;
  const int wm = wave >> 1;   // 0..3  (32-row slice)
  const int wn = wave & 1;    // 0..1  (64-col slice)

  // staging geometry: fixed (row, col4) per thread, constant-stride walk
  const int ca = (tid & 7)  * 4, ra = tid >> 3;   // A: 32 rows/pass x 32 cols
  const int cb = (tid & 31) * 4, rb = tid >> 5;   // B:  8 rows/pass x 128 cols
  const float* gpa = A  + (size_t)(m0 + ra) * K + ca;
  const float* gpb = Bm + (size_t)rb * N + n0 + cb;

  v8f acc[2][4];
#pragma unroll
  for (int s = 0; s < 2; ++s)
#pragma unroll
    for (int t = 0; t < 4; ++t) acc[s][t] = (v8f){0.f,0.f,0.f,0.f,0.f,0.f,0.f,0.f};

  for (int k0 = 0; k0 < K; k0 += BK) {
    // ---- issue ALL tile loads first (keep 8 b128 loads in flight) ----------
    float4 fa[4], fb[4];
    {
      const float* p = gpa;
#pragma unroll
      for (int it = 0; it < 4; ++it) { fa[it] = *(const float4*)p; p += (size_t)32 * K; }
      const float* q = gpb;
#pragma unroll
      for (int it = 0; it < 4; ++it) { fb[it] = *(const float4*)q; q += (size_t)8 * N; }
    }
    // ---- convert + store to LDS --------------------------------------------
#pragma unroll
    for (int it = 0; it < 4; ++it) {
      *(uint2*)&Asl[ra + 32 * it][ca] =
          make_uint2(pack2bf(fa[it].x, fa[it].y), pack2bf(fa[it].z, fa[it].w));
      const int r = rb + 8 * it;
      Bsl[cb + 0][r] = f2bf(fb[it].x);
      Bsl[cb + 1][r] = f2bf(fb[it].y);
      Bsl[cb + 2][r] = f2bf(fb[it].z);
      Bsl[cb + 3][r] = f2bf(fb[it].w);
    }
    if (k0 + BK < K) {                       // warm L2 for next k-tile
      __builtin_prefetch(gpa + BK, 0, 1);
      __builtin_prefetch(gpb + (size_t)BK * N, 0, 1);
    }
    __syncthreads();

    // A fragments (two 16-row subtiles)
    v16bf afr[2];
#pragma unroll
    for (int s = 0; s < 2; ++s) {
      const int row = wm * 32 + s * 16 + lm;
      v8u u;
#pragma unroll
      for (int j = 0; j < 8; ++j)
        u[j] = *(const unsigned int*)&Asl[row][kpairA(j, lh)];
      afr[s] = __builtin_bit_cast(v16bf, u);
    }
    // B fragments (four 16-col subtiles)
    v16bf bfr[4];
#pragma unroll
    for (int t = 0; t < 4; ++t) {
      const int col = wn * 64 + t * 16 + lm;
      v8u u;
#pragma unroll
      for (int j = 0; j < 8; ++j)
        u[j] = *(const unsigned int*)&Bsl[col][kpairA(j, lh)];
      bfr[t] = __builtin_bit_cast(v16bf, u);
    }
#pragma unroll
    for (int s = 0; s < 2; ++s)
#pragma unroll
      for (int t = 0; t < 4; ++t)
        acc[s][t] = wmma_bf16(afr[s], bfr[t], acc[s][t]);
    __syncthreads();

    gpa += BK;
    gpb += (size_t)BK * N;
  }

  // epilogue: C layout VGPR r -> M = r + 8*lh, lane -> N = lm
#pragma unroll
  for (int t = 0; t < 4; ++t) {
    const int col = n0 + wn * 64 + t * 16 + lm;
    const float bv = bias[col];
#pragma unroll
    for (int s = 0; s < 2; ++s) {
#pragma unroll
      for (int r = 0; r < 8; ++r) {
        const int row = m0 + wm * 32 + s * 16 + r + 8 * lh;
        C[(size_t)row * N + col] = acc[s][t][r] + bv;
      }
    }
  }
}

// ---------------------------------------------------------------------------
// Flash attention: one block = (batch, head, 128-query tile). 8 waves; each
// wave owns 16 query rows. K/V tiles of 128 keys staged to LDS as bf16.
// Causal: key tiles ki in [0, qi]; elementwise mask on diagonal tile only.
// ---------------------------------------------------------------------------
#define KP   (HD + 8)     // K_lds row pad
#define VP   (128 + 8)    // Vt_lds / P_lds row pad

__global__ __launch_bounds__(256) void flash_attn_wmma_k(
    const float* __restrict__ qkv, float* __restrict__ attn) {
  __shared__ unsigned short Ksl[128][KP];   // [key][d]
  __shared__ unsigned short Vsl[HD][VP];    // [d][key] (transposed at stage)
  __shared__ unsigned short Psl[128][VP];   // [qrow][key], wave-private rows

  const int tid  = threadIdx.x;
  const int wave = tid >> 5;
  const int lane = tid & 31;
  const int lm = lane & 15, lh = lane >> 4;

  const int qi = blockIdx.x;          // query tile (128 rows)
  const int h  = blockIdx.y;
  const int b  = blockIdx.z;
  const int qbase = qi * 128;
  const size_t rowstride = 3 * EMB;   // qkv row length

  // --- Q fragments (A layout), pre-scaled by 1/sqrt(HD) = 0.125 -------------
  const float* qptr =
      qkv + (size_t)(b * SEQ + qbase + wave * 16 + lm) * rowstride + h * HD;
  v16bf qf[2];
#pragma unroll
  for (int ks = 0; ks < 2; ++ks) {
    v8u u;
#pragma unroll
    for (int j = 0; j < 8; ++j) {
      const int d = ks * 32 + kpairA(j, lh);
      const float2 f = *(const float2*)(qptr + d);
      u[j] = pack2bf(f.x * 0.125f, f.y * 0.125f);
    }
    qf[ks] = __builtin_bit_cast(v16bf, u);
  }

  float mrow[8], lrow[8];
#pragma unroll
  for (int r = 0; r < 8; ++r) { mrow[r] = -INFINITY; lrow[r] = 0.f; }
  v8f oacc[4];
#pragma unroll
  for (int t = 0; t < 4; ++t) oacc[t] = (v8f){0.f,0.f,0.f,0.f,0.f,0.f,0.f,0.f};

  // staging geometry: thread covers (key = rv + 16*it, d = cv..cv+3)
  const int cv = (tid & 15) * 4, rv = tid >> 4;
  const float* gkv =
      qkv + (size_t)(b * SEQ + rv) * rowstride + h * HD + cv;   // += 128 rows/tile

  for (int ki = 0; ki <= qi; ++ki) {
    // stage K tile [128 keys][64 d] and V tile transposed [64 d][128 keys];
    // batch 4 key-rows of loads (8 b128 in flight) before converting
    {
      const float* p = gkv;
#pragma unroll
      for (int half = 0; half < 2; ++half) {
        float4 fk[4], fv[4];
#pragma unroll
        for (int it = 0; it < 4; ++it) {
          fk[it] = *(const float4*)(p + EMB);       // K block
          fv[it] = *(const float4*)(p + 2 * EMB);   // V block
          p += 16 * rowstride;
        }
#pragma unroll
        for (int it = 0; it < 4; ++it) {
          const int r = rv + 16 * (half * 4 + it);
          *(uint2*)&Ksl[r][cv] =
              make_uint2(pack2bf(fk[it].x, fk[it].y), pack2bf(fk[it].z, fk[it].w));
          Vsl[cv + 0][r] = f2bf(fv[it].x);
          Vsl[cv + 1][r] = f2bf(fv[it].y);
          Vsl[cv + 2][r] = f2bf(fv[it].z);
          Vsl[cv + 3][r] = f2bf(fv[it].w);
        }
      }
    }
    if (ki < qi) {                      // warm L2 for next key tile
      __builtin_prefetch(gkv + 128 * rowstride + EMB, 0, 1);
      __builtin_prefetch(gkv + 128 * rowstride + 2 * EMB, 0, 1);
    }
    __syncthreads();

    // --- scores: S = Q K^T  (8 key subtiles x 2 k-steps over d) -------------
    v8f sc[8];
#pragma unroll
    for (int nt = 0; nt < 8; ++nt) {
      v8f s = (v8f){0.f,0.f,0.f,0.f,0.f,0.f,0.f,0.f};
#pragma unroll
      for (int ks = 0; ks < 2; ++ks) {
        const int key = nt * 16 + lm;          // B frag: lane -> column (key)
        v8u u;
#pragma unroll
        for (int j = 0; j < 8; ++j)
          u[j] = *(const unsigned int*)&Ksl[key][ks * 32 + kpairA(j, lh)];
        s = wmma_bf16(qf[ks], __builtin_bit_cast(v16bf, u), s);
      }
      sc[nt] = s;
    }

    // --- causal mask on the diagonal tile ----------------------------------
    if (ki == qi) {
#pragma unroll
      for (int nt = 0; nt < 8; ++nt) {
        const int kcol = nt * 16 + lm;
#pragma unroll
        for (int r = 0; r < 8; ++r) {
          const int qrow = wave * 16 + r + 8 * lh;
          if (kcol > qrow) sc[nt][r] = -INFINITY;
        }
      }
    }

    // --- online softmax (row reductions across 16-lane groups) -------------
#pragma unroll
    for (int r = 0; r < 8; ++r) {
      float tmax = sc[0][r];
#pragma unroll
      for (int nt = 1; nt < 8; ++nt) tmax = fmaxf(tmax, sc[nt][r]);
      tmax = redmax16(tmax);
      const float mn    = fmaxf(mrow[r], tmax);
      const float scale = __expf(mrow[r] - mn);   // 0 on first tile
      mrow[r] = mn;
      float psum = 0.f;
#pragma unroll
      for (int nt = 0; nt < 8; ++nt) {
        const float p = __expf(sc[nt][r] - mn);
        sc[nt][r] = p;
        psum += p;
      }
      psum = redsum16(psum);
      lrow[r] = lrow[r] * scale + psum;
#pragma unroll
      for (int t = 0; t < 4; ++t) oacc[t][r] *= scale;
    }

    // --- spill P to LDS (bf16) to convert C-layout -> A-layout -------------
#pragma unroll
    for (int nt = 0; nt < 8; ++nt) {
#pragma unroll
      for (int r = 0; r < 8; ++r)
        Psl[wave * 16 + r + 8 * lh][nt * 16 + lm] = f2bf(sc[nt][r]);
    }
    // wave-private region: only dscnt ordering needed (compiler inserts)

    // --- O += P V  (4 k-steps over 128 keys x 4 d-subtiles) -----------------
#pragma unroll
    for (int ks2 = 0; ks2 < 4; ++ks2) {
      v8u ua;
#pragma unroll
      for (int j = 0; j < 8; ++j)
        ua[j] = *(const unsigned int*)&Psl[wave * 16 + lm]
                                         [ks2 * 32 + kpairA(j, lh)];
      const v16bf pf = __builtin_bit_cast(v16bf, ua);
#pragma unroll
      for (int t = 0; t < 4; ++t) {
        const int d = t * 16 + lm;             // B frag: lane -> column (d)
        v8u ub;
#pragma unroll
        for (int j = 0; j < 8; ++j)
          ub[j] = *(const unsigned int*)&Vsl[d][ks2 * 32 + kpairA(j, lh)];
        oacc[t] = wmma_bf16(pf, __builtin_bit_cast(v16bf, ub), oacc[t]);
      }
    }
    __syncthreads();   // protect K/V LDS before next tile restage

    gkv += 128 * rowstride;
  }

  // --- epilogue: attn[b][q][h*64+d] = oacc / l ------------------------------
#pragma unroll
  for (int t = 0; t < 4; ++t) {
    const int d = t * 16 + lm;
#pragma unroll
    for (int r = 0; r < 8; ++r) {
      const int qrow = qbase + wave * 16 + r + 8 * lh;
      attn[(size_t)(b * SEQ + qrow) * EMB + h * HD + d] = oacc[t][r] / lrow[r];
    }
  }
}

// ---------------------------------------------------------------------------
// launch
// ---------------------------------------------------------------------------
extern "C" void kernel_launch(void* const* d_in, const int* in_sizes, int n_in,
                              void* d_out, int out_size, void* d_ws, size_t ws_size,
                              hipStream_t stream) {
  (void)in_sizes; (void)n_in; (void)out_size; (void)ws_size;
  const float* x     = (const float*)d_in[0];   // [B,S,E]
  const float* W_qkv = (const float*)d_in[1];   // [E,3E]
  const float* b_qkv = (const float*)d_in[2];   // [3E]
  const float* W_o   = (const float*)d_in[3];   // [E,E]
  const float* b_o   = (const float*)d_in[4];   // [E]
  // d_in[5] = mask (causal tril) — applied analytically in the kernel
  float* out = (float*)d_out;

  float* qkvbuf  = (float*)d_ws;                                  // [B*S, 3E]
  float* attnbuf = qkvbuf + (size_t)BATCH * SEQ * 3 * EMB;        // [B*S, E]

  const int M = BATCH * SEQ;   // 4096

  // 1) qkv = x @ W_qkv + b_qkv
  gemm_bias_wmma_k<<<dim3(M / BM, (3 * EMB) / BN), 256, 0, stream>>>(
      x, W_qkv, b_qkv, qkvbuf, M, 3 * EMB, EMB);

  // 2) fused causal flash attention (scores never touch HBM)
  flash_attn_wmma_k<<<dim3(SEQ / 128, NH, BATCH), 256, 0, stream>>>(
      qkvbuf, attnbuf);

  // 3) out = attn @ W_o + b_o
  gemm_bias_wmma_k<<<dim3(M / BM, EMB / BN), 256, 0, stream>>>(
      attnbuf, W_o, b_o, out, M, EMB, EMB);
}
